// Attention_37709812859374
// MI455X (gfx1250) — compile-verified, matrix operand order checked by reading, not run on hardware
//
#include <hip/hip_runtime.h>
#include <hip/hip_bf16.h>

// ---------------------------------------------------------------------------
// MI455X (gfx1250, wave32) local-attention pipeline, all matmuls on
// v_wmma_f32_16x16x32_bf16, with async global->LDS copies (ASYNCcnt) for the
// bf16 Q/K tiles in the attention kernel and global_prefetch for the bias
// window.
//
//   1) gemm64<EPI_HS_BF16>   : Q/K/V projections -> bf16, head-split [H][N][D]
//   2) gemm64<EPI_SIG_F32>   : gate G = sigmoid(q_x @ Wg)            (f32)
//   3) local_attn            : per (head, trunk) S=QK^T+bias, softmax, O=PV
//   4) gemm64<EPI_F32,GATED> : out = (O * G) @ Wo                    (f32)
// ---------------------------------------------------------------------------

typedef __bf16 bf16;
typedef __attribute__((ext_vector_type(16))) __bf16 v16bf;
typedef __attribute__((ext_vector_type(8)))  __bf16 bh8;
typedef __attribute__((ext_vector_type(8)))  float  v8f;
typedef __attribute__((ext_vector_type(4)))  int    v4i;

#define V8F_ZERO (v8f){0.f,0.f,0.f,0.f,0.f,0.f,0.f,0.f}

#if __has_builtin(__builtin_amdgcn_global_load_async_to_lds_b128) && \
    __has_builtin(__builtin_amdgcn_s_wait_asynccnt)
#define HAVE_ASYNC_LDS 1
typedef __attribute__((address_space(1))) v4i* gv4i_p;   // global int4*
typedef __attribute__((address_space(3))) v4i* lv4i_p;   // LDS    int4*
#else
#define HAVE_ASYNC_LDS 0
#endif

union FragU { v16bf v; bh8 h[2]; };

// WMMA bf16 A/B fragment layout (wave32, 16x32 A or 32x16 B):
//   element index e (0..15), lane l:
//     M_or_N = l & 15
//     K      = (e & 7) | ((e & 8) << 1) | ((l & 16) >> 1)
// => per lane: two contiguous runs of 8 halves at K-offsets {base, base+16},
//    base = (l&16)>>1.  Load as two 16-byte ds_load_b128.
__device__ __forceinline__ v16bf ldsFrag(const bf16* p) {
    FragU f;
    f.h[0] = *(const bh8*)(p);
    f.h[1] = *(const bh8*)(p + 16);
    return f.v;
}

// C/D f32 16x16 layout: element i (0..7), lane l:  M = i + 8*(l>>4), N = l&15.

enum { EPI_F32 = 0, EPI_HS_BF16 = 1, EPI_SIG_F32 = 2 };

// ---------------------------------------------------------------------------
// Generic 64x64-tile bf16-WMMA GEMM:  Out = epilogue( (A [*Gate]) @ B * scale )
//   A: [M][K] f32 (row major), B: [K][N] f32, K multiple of 256.
//   4 waves / block, wave w owns rows [16w,16w+16) of the tile.
//   (A/B arrive as f32 and are converted to bf16 in the VALU on the way into
//    LDS, so this path keeps the synchronous load+ds_store pipeline.)
// ---------------------------------------------------------------------------
template <int MODE, bool GATED>
__global__ __launch_bounds__(128) void gemm64(
    const float* __restrict__ A, const float* __restrict__ B,
    const float* __restrict__ Gate, void* __restrict__ Out,
    int M, int N, int K, float scale)
{
    constexpr int AS = 264;                 // row stride (halves), odd*16B: no bank conflicts
    constexpr int BS = 264;
    __shared__ bf16 As [64 * AS];           // A tile  64 x 256   (row major)
    __shared__ bf16 BsT[64 * BS];           // B tile  256 x 64, stored transposed [n][k]

    const int tid = threadIdx.x;
    const int l   = tid & 31;
    const int w   = tid >> 5;
    const int m0  = blockIdx.y * 64;
    const int n0  = blockIdx.x * 64;
    const int fb  = (l & 16) >> 1;          // fragment K base offset per lane half

    v8f acc[4];
#pragma unroll
    for (int ct = 0; ct < 4; ct++) acc[ct] = V8F_ZERO;

    for (int k0 = 0; k0 < K; k0 += 256) {
        __syncthreads();
        // A tile (coalesced over k), fused elementwise gate for the final GEMM
        for (int idx = tid; idx < 64 * 256; idx += 128) {
            int r = idx >> 8, c = idx & 255;
            float a = A[(size_t)(m0 + r) * K + (k0 + c)];
            if (GATED) a *= Gate[(size_t)(m0 + r) * K + (k0 + c)];
            As[r * AS + c] = (bf16)a;
        }
        // B tile (coalesced over n), transposed into LDS so B-fragments are contiguous
        for (int idx = tid; idx < 256 * 64; idx += 128) {
            int r = idx >> 6, c = idx & 63;
            BsT[c * BS + r] = (bf16)B[(size_t)(k0 + r) * N + (n0 + c)];
        }
        __syncthreads();

#pragma unroll
        for (int kc = 0; kc < 256; kc += 32) {
            v16bf a = ldsFrag(&As[(16 * w + (l & 15)) * AS + kc + fb]);
#pragma unroll
            for (int ct = 0; ct < 4; ct++) {
                v16bf b = ldsFrag(&BsT[(16 * ct + (l & 15)) * BS + kc + fb]);
                acc[ct] = __builtin_amdgcn_wmma_f32_16x16x32_bf16(
                    false, a, false, b, (short)0, acc[ct], false, false);
            }
        }
    }

#pragma unroll
    for (int ct = 0; ct < 4; ct++) {
#pragma unroll
        for (int i = 0; i < 8; i++) {
            int gm = m0 + 16 * w + i + ((l >> 4) << 3);
            int gn = n0 + 16 * ct + (l & 15);
            float x = acc[ct][i] * scale;
            if (MODE == EPI_F32) {
                ((float*)Out)[(size_t)gm * N + gn] = x;
            } else if (MODE == EPI_SIG_F32) {
                ((float*)Out)[(size_t)gm * N + gn] = 1.f / (1.f + __expf(-x));
            } else { // EPI_HS_BF16: head-split [H][M][64] bf16
                int hh = gn >> 6, dd = gn & 63;
                ((bf16*)Out)[((size_t)hh * M + gm) * 64 + dd] = (bf16)x;
            }
        }
    }
}

// ---------------------------------------------------------------------------
// Local attention, one (head, trunk) per 128-thread block.
//   Trunk t: queries [t*64, t*64+64), keys at indices t*64-96 + [0,256).
//   OOB keys/values/bias are ZERO (not -inf) => scores exactly 0 there,
//   and softmax runs over all 256 columns, matching the reference.
//   Q/K tiles are bf16 in global and land unchanged in LDS -> async copies.
// ---------------------------------------------------------------------------
__global__ __launch_bounds__(128) void local_attn(
    const bf16* __restrict__ Qh, const bf16* __restrict__ Kh,
    const bf16* __restrict__ Vh, const float* __restrict__ bias,
    float* __restrict__ Op, int N)
{
    constexpr int QS = 72, KS = 72, VS = 264, PS = 264;
    __shared__ bf16 Qs [64  * QS];          // Q  [q][d]
    __shared__ bf16 Ks [256 * KS];          // K  [key][d]  (B-frag of K^T contiguous in d)
    __shared__ bf16 VsT[64  * VS];          // V^T[d][key]  (B-frag of V contiguous in key)
    __shared__ bf16 Ps [64  * PS];          // P  [q][key]

    const int tid = threadIdx.x;
    const int l   = tid & 31;
    const int w   = tid >> 5;
    const int T   = N >> 6;
    const int t   = blockIdx.x % T;
    const int h   = blockIdx.x / T;
    const int kk0 = t * 64 - 96;            // pad_left = (256-64)/2
    const int fb  = (l & 16) >> 1;

    // ---- Q tile: 64 rows x 8 x16B chunks, always in range ----
#if HAVE_ASYNC_LDS
    for (int idx = tid; idx < 64 * 8; idx += 128) {
        int r = idx >> 3, c8 = (idx & 7) * 8;
        __builtin_amdgcn_global_load_async_to_lds_b128(
            (gv4i_p)(Qh + ((size_t)h * N + t * 64 + r) * 64 + c8),
            (lv4i_p)(&Qs[r * QS + c8]),
            0, 0);
    }
#else
    for (int idx = tid; idx < 64 * 64; idx += 128) {
        int r = idx >> 6, c = idx & 63;
        Qs[r * QS + c] = Qh[((size_t)h * N + t * 64 + r) * 64 + c];
    }
#endif

    // ---- K tile: 256 rows x 8 x16B chunks; async in-range, ds-zero OOB ----
#if HAVE_ASYNC_LDS
    for (int idx = tid; idx < 256 * 8; idx += 128) {
        int r = idx >> 3, c8 = (idx & 7) * 8;
        int kk = kk0 + r;
        if (kk >= 0 && kk < N) {
            __builtin_amdgcn_global_load_async_to_lds_b128(
                (gv4i_p)(Kh + ((size_t)h * N + kk) * 64 + c8),
                (lv4i_p)(&Ks[r * KS + c8]),
                0, 0);
        } else {
            *(bh8*)&Ks[r * KS + c8] = (bh8){(bf16)0.f, (bf16)0.f, (bf16)0.f, (bf16)0.f,
                                            (bf16)0.f, (bf16)0.f, (bf16)0.f, (bf16)0.f};
        }
    }
#endif

    // ---- prefetch the bias window for this trunk (overlaps LDS fill / QK^T) ----
    {
        int lo = (kk0 > 0) ? kk0 : 0;
        int hi = (kk0 + 256 < N) ? (kk0 + 256) : N;
        for (int idx = tid; idx < 64 * 8; idx += 128) {   // 64 rows x up to 8 cachelines
            int r   = t * 64 + (idx >> 3);
            int col = lo + (idx & 7) * 32;                // 32 f32 = 128 B
            if (col < hi) __builtin_prefetch(&bias[(size_t)r * N + col], 0, 3);
        }
    }

    // ---- V tile (transposed into LDS -> must go through VGPRs) + sync-K fallback ----
    for (int idx = tid; idx < 256 * 64; idx += 128) {
        int r = idx >> 6, c = idx & 63;
        int kk = kk0 + r;
        bf16 vv = (bf16)0.f;
        if (kk >= 0 && kk < N) vv = Vh[((size_t)h * N + kk) * 64 + c];
        VsT[c * VS + r] = vv;
#if !HAVE_ASYNC_LDS
        bf16 kv = (bf16)0.f;
        if (kk >= 0 && kk < N) kv = Kh[((size_t)h * N + kk) * 64 + c];
        Ks[r * KS + c] = kv;
#endif
    }

#if HAVE_ASYNC_LDS
    __builtin_amdgcn_s_wait_asynccnt(0);
#endif
    __syncthreads();

    // ---- S = Q K^T (rows 16w..16w+15, 16 column tiles of 16) ----
    v8f s[16];
#pragma unroll
    for (int ct = 0; ct < 16; ct++) s[ct] = V8F_ZERO;
#pragma unroll
    for (int kc = 0; kc < 64; kc += 32) {
        v16bf a = ldsFrag(&Qs[(16 * w + (l & 15)) * QS + kc + fb]);
#pragma unroll
        for (int ct = 0; ct < 16; ct++) {
            v16bf b = ldsFrag(&Ks[(16 * ct + (l & 15)) * KS + kc + fb]);
            s[ct] = __builtin_amdgcn_wmma_f32_16x16x32_bf16(
                false, a, false, b, (short)0, s[ct], false, false);
        }
    }

    // ---- + bias (only in-range keys; OOB bias is zero-padded) ----
    const int rbase = t * 64 + 16 * w + ((l >> 4) << 3);
#pragma unroll
    for (int ct = 0; ct < 16; ct++) {
        int key = 16 * ct + (l & 15);
        int kk  = kk0 + key;
        if (kk >= 0 && kk < N) {
#pragma unroll
            for (int i = 0; i < 8; i++)
                s[ct][i] += bias[(size_t)(rbase + i) * N + kk];
        }
    }

    // ---- softmax over 256 columns (row = (i, lane-halfgroup); N = lane&15) ----
#pragma unroll
    for (int i = 0; i < 8; i++) {
        float mx = s[0][i];
#pragma unroll
        for (int ct = 1; ct < 16; ct++) mx = fmaxf(mx, s[ct][i]);
#pragma unroll
        for (int d = 1; d < 16; d <<= 1) mx = fmaxf(mx, __shfl_xor(mx, d, 32));
        float sum = 0.f;
#pragma unroll
        for (int ct = 0; ct < 16; ct++) {
            float p = __expf(s[ct][i] - mx);
            s[ct][i] = p;
            sum += p;
        }
#pragma unroll
        for (int d = 1; d < 16; d <<= 1) sum += __shfl_xor(sum, d, 32);
        float inv = 1.f / sum;
#pragma unroll
        for (int ct = 0; ct < 16; ct++) s[ct][i] *= inv;
    }

    // ---- P -> LDS (bf16) ----
#pragma unroll
    for (int ct = 0; ct < 16; ct++)
#pragma unroll
        for (int i = 0; i < 8; i++)
            Ps[(16 * w + i + ((l >> 4) << 3)) * PS + 16 * ct + (l & 15)] =
                (bf16)s[ct][i];
    __syncthreads();

    // ---- O = P V  (4 tiles over D) ----
    v8f o[4];
#pragma unroll
    for (int ct = 0; ct < 4; ct++) o[ct] = V8F_ZERO;
#pragma unroll
    for (int kc = 0; kc < 256; kc += 32) {
        v16bf a = ldsFrag(&Ps[(16 * w + (l & 15)) * PS + kc + fb]);
#pragma unroll
        for (int ct = 0; ct < 4; ct++) {
            v16bf b = ldsFrag(&VsT[(16 * ct + (l & 15)) * VS + kc + fb]);
            o[ct] = __builtin_amdgcn_wmma_f32_16x16x32_bf16(
                false, a, false, b, (short)0, o[ct], false, false);
        }
    }

    // ---- write O merged back to [N][H*64] f32 ----
#pragma unroll
    for (int ct = 0; ct < 4; ct++)
#pragma unroll
        for (int i = 0; i < 8; i++) {
            int gm = t * 64 + 16 * w + i + ((l >> 4) << 3);
            int gn = h * 64 + 16 * ct + (l & 15);
            Op[(size_t)gm * 1024 + gn] = o[ct][i];
        }
}

// ---------------------------------------------------------------------------
extern "C" void kernel_launch(void* const* d_in, const int* in_sizes, int n_in,
                              void* d_out, int out_size, void* d_ws, size_t ws_size,
                              hipStream_t stream) {
    const float* q_x  = (const float*)d_in[0];   // [1,4096,256]
    const float* kv_x = (const float*)d_in[1];   // [1,4096,256]
    const float* bias = (const float*)d_in[2];   // [1,1,4096,4096]
    const float* Wq   = (const float*)d_in[3];   // [256,1024]
    const float* Wk   = (const float*)d_in[4];
    const float* Wv   = (const float*)d_in[5];
    const float* Wg   = (const float*)d_in[6];
    const float* Wo   = (const float*)d_in[7];   // [1024,256]
    // d_in[8]=n_queries(64), d_in[9]=n_keys(256): fixed by setup, hardcoded.

    const int Nn = 4096, C = 256, HD = 1024;

    char*  ws = (char*)d_ws;
    bf16*  Qh = (bf16*)(ws);                          //  8 MB  [16][4096][64] bf16
    bf16*  Kh = (bf16*)(ws + ((size_t)8  << 20));     //  8 MB
    bf16*  Vh = (bf16*)(ws + ((size_t)16 << 20));     //  8 MB
    float* G  = (float*)(ws + ((size_t)24 << 20));    // 16 MB  [4096][1024] f32
    float* Op = (float*)(ws + ((size_t)40 << 20));    // 16 MB  [4096][1024] f32

    dim3 blk(128);
    dim3 gproj(HD / 64, Nn / 64);

    // Q (pre-scaled by 1/sqrt(64)), K, V  -> head-split bf16
    gemm64<EPI_HS_BF16, false><<<gproj, blk, 0, stream>>>(q_x,  Wq, nullptr, Qh, Nn, HD, C, 0.125f);
    gemm64<EPI_HS_BF16, false><<<gproj, blk, 0, stream>>>(kv_x, Wk, nullptr, Kh, Nn, HD, C, 1.f);
    gemm64<EPI_HS_BF16, false><<<gproj, blk, 0, stream>>>(kv_x, Wv, nullptr, Vh, Nn, HD, C, 1.f);
    // gate
    gemm64<EPI_SIG_F32, false><<<gproj, blk, 0, stream>>>(q_x,  Wg, nullptr, G,  Nn, HD, C, 1.f);
    // attention: 16 heads x 64 trunks
    local_attn<<<dim3(16 * (Nn / 64)), blk, 0, stream>>>(Qh, Kh, Vh, bias, Op, Nn);
    // out = (O * G) @ Wo
    gemm64<EPI_F32, true><<<dim3(C / 64, Nn / 64), blk, 0, stream>>>(Op, Wo, G, d_out, Nn, C, HD, 1.f);
}
